// KeywordSpottingModel_32658931319052
// MI455X (gfx1250) — compile-verified, matrix operand order checked by reading, not run on hardware
//
#include <hip/hip_runtime.h>
#include <hip/hip_bf16.h>

// ---------------------------------------------------------------------------
// Mamba keyword-spotting forward pass for MI455X (gfx1250, wave32, WMMA).
//
// Sizes: B=16, T=256, IN_DIM=80, D_MODEL=256, D_INNER=512, D_STATE=16,
//        D_CONV=4, DT_RANK=16, N_LAYERS=4, N_LABELS=35.  Seq axis L == 256.
//
// GEMMs: v_wmma_f32_16x16x32_bf16, f32 accumulate, bf16 operands staged with
// K % 32 == 0.  Inner loop is register double-buffered: loads for K-step
// kb overlap the 4 wmma of K-step kb-32.  Scan / conv / norms stay f32 VALU.
// ---------------------------------------------------------------------------

typedef __attribute__((ext_vector_type(16))) __bf16 v16bf;
typedef __attribute__((ext_vector_type(8)))  __bf16 v8bf;
typedef __attribute__((ext_vector_type(8)))  float  v8f;

static __device__ __forceinline__ __bf16 f2bf(float f) {
  unsigned u = __builtin_bit_cast(unsigned, f);
  unsigned r = u + 0x7FFFu + ((u >> 16) & 1u);       // round-to-nearest-even
  unsigned short h = (unsigned short)(r >> 16);
  return __builtin_bit_cast(__bf16, h);
}

static __device__ __forceinline__ float silu_f(float x) {
  return x / (1.f + __expf(-x));
}

static __device__ __forceinline__ v16bf cat16(v8bf lo, v8bf hi) {
  return __builtin_shufflevector(lo, hi,
      0, 1, 2, 3, 4, 5, 6, 7, 8, 9, 10, 11, 12, 13, 14, 15);
}

// ---------------------------------------------------------------------------
// WMMA GEMM:  C[M,N] = act(A[M,K] * W[N,K]^T + bias[N])
//   A, W: bf16, rows 16B-aligned, K % 32 == 0 (caller zero-pads).
//   block = 128 threads = 4 waves; wave owns a 16x64 tile -> 4 wmma / K-step,
//   A fragment reused.  Lane's 16 K-values = two contiguous 8-elem spans
//   ([kb+8g, +8) and [kb+16+8g, +8)) -> two global_load_b128 per fragment.
//   Register double-buffering: loads of step kb issued before wmma of step
//   kb-32 so load latency overlaps the matrix pipe.
//   ACT: 0 = none, 1 = softplus.
//   TRANS_BF16_OUT: store bf16 at out[b*65536 + n*256 + t] (m = b*256+t);
//   writes the next GEMM's operand layout directly (T = 256).
//   GUARD: enable M/N store masking (only needed for N=48 / fc tiles).
// ---------------------------------------------------------------------------
template <int ACT, bool TRANS_BF16_OUT, bool GUARD>
__global__ __launch_bounds__(128)
void wmma_gemm_kernel(const __bf16* __restrict__ A, int lda,
                      const __bf16* __restrict__ W, int ldw,
                      const float* __restrict__ bias,
                      void* __restrict__ Cv, int ldc,
                      int M, int N, int K) {
  const int lane  = threadIdx.x & 31;
  const int wave  = threadIdx.x >> 5;
  const int tileM = blockIdx.y * 64 + wave * 16;
  const int tileN = blockIdx.x * 64;
  const int g     = lane >> 4;      // K half-group select (ISA A/B layout)
  const int r16   = lane & 15;

  int m = tileM + r16;
  if (GUARD && m >= M) m = M - 1;   // clamp; OOB rows masked at store
  const __bf16* Arow = A + (size_t)m * lda + (g << 3);

  const __bf16* Wrow[4];
#pragma unroll
  for (int j = 0; j < 4; ++j) {
    int n = tileN + j * 16 + r16;
    if (GUARD && n >= N) n = N - 1; // clamp; OOB cols masked at store
    Wrow[j] = W + (size_t)n * ldw + (g << 3);
  }

  v8f acc[4] = {};

  // ---- prologue: load K-step 0 ----
  v8bf alo = *(const v8bf*)(Arow);
  v8bf ahi = *(const v8bf*)(Arow + 16);
  v8bf blo[4], bhi[4];
#pragma unroll
  for (int j = 0; j < 4; ++j) {
    blo[j] = *(const v8bf*)(Wrow[j]);
    bhi[j] = *(const v8bf*)(Wrow[j] + 16);
  }

  // ---- pipelined main loop ----
  for (int kb = 32; kb < K; kb += 32) {
    v8bf nalo = *(const v8bf*)(Arow + kb);
    v8bf nahi = *(const v8bf*)(Arow + kb + 16);
    v8bf nblo[4], nbhi[4];
#pragma unroll
    for (int j = 0; j < 4; ++j) {
      nblo[j] = *(const v8bf*)(Wrow[j] + kb);
      nbhi[j] = *(const v8bf*)(Wrow[j] + kb + 16);
    }
    v16bf af = cat16(alo, ahi);
#pragma unroll
    for (int j = 0; j < 4; ++j)
      acc[j] = __builtin_amdgcn_wmma_f32_16x16x32_bf16(
          false, af, false, cat16(blo[j], bhi[j]), (short)0, acc[j],
          false, false);
    alo = nalo; ahi = nahi;
#pragma unroll
    for (int j = 0; j < 4; ++j) { blo[j] = nblo[j]; bhi[j] = nbhi[j]; }
  }

  // ---- drain last K-step ----
  {
    v16bf af = cat16(alo, ahi);
#pragma unroll
    for (int j = 0; j < 4; ++j)
      acc[j] = __builtin_amdgcn_wmma_f32_16x16x32_bf16(
          false, af, false, cat16(blo[j], bhi[j]), (short)0, acc[j],
          false, false);
  }

  // ---- epilogue: C/D layout -> vgpr r holds M = tileM + r + 8*(lane>>4) ----
#pragma unroll
  for (int j = 0; j < 4; ++j) {
    int n = tileN + j * 16 + r16;
#pragma unroll
    for (int rr = 0; rr < 8; ++rr) {
      int mm = tileM + rr + (g << 3);
      if (!GUARD || (mm < M && n < N)) {
        float v = acc[j][rr];
        if (bias) v += bias[n];
        if (ACT == 1) v = (v > 20.f) ? v : __logf(1.f + __expf(v));  // softplus
        if (TRANS_BF16_OUT) {
          ((__bf16*)Cv)[(((size_t)(mm >> 8)) << 16) + ((size_t)n << 8) +
                        (mm & 255)] = f2bf(v);
        } else {
          ((float*)Cv)[(size_t)mm * ldc + n] = v;
        }
      }
    }
  }
}

// ---------------------------------------------------------------------------
// Generic f32 -> bf16 staging copy with zero K-padding:
//   dst[r*ldd + c] = (c < copyC) ? bf16(src[r*lds + c]) : 0
// ---------------------------------------------------------------------------
__global__ void cvt_pad_bf16_kernel(const float* __restrict__ src, int lds,
                                    int copyC, __bf16* __restrict__ dst,
                                    int ldd, int total) {
  int idx = blockIdx.x * blockDim.x + threadIdx.x;
  if (idx >= total) return;
  int c = idx % ldd;
  int r = idx / ldd;
  float v = (c < copyC) ? src[(size_t)r * lds + c] : 0.f;
  dst[idx] = f2bf(v);
}

// ---------------------------------------------------------------------------
// xT[m = b*256+t][k] = bf16(x[b][k][t]) for k<80, zero-pad to K=96.
// ---------------------------------------------------------------------------
__global__ void transpose_x_kernel(const float* __restrict__ x,
                                   __bf16* __restrict__ xT) {
  int idx = blockIdx.x * blockDim.x + threadIdx.x;
  if (idx >= 16 * 256 * 96) return;
  int k = idx % 96;
  int m = idx / 96;
  int t = m & 255, b = m >> 8;
  float v = (k < 80) ? x[((size_t)b * 80 + k) * 256 + t] : 0.f;
  xT[idx] = f2bf(v);
}

// ---------------------------------------------------------------------------
// Depthwise causal conv (width 4) + bias + silu.  Branchless.
// Writes f32 (scan input) and bf16 (x_proj GEMM operand).
// ---------------------------------------------------------------------------
__global__ void conv_silu_kernel(const float* __restrict__ xz,
                                 const float* __restrict__ cw,   // [512,4]
                                 const float* __restrict__ cb,   // [512]
                                 float* __restrict__ xcp,
                                 __bf16* __restrict__ xcp_bf) {
  int idx = blockIdx.x * blockDim.x + threadIdx.x;   // B*L*512
  if (idx >= 16 * 256 * 512) return;
  int d = idx & 511;
  int m = idx >> 9;          // b*256 + l
  int l = m & 255;
  int b = m >> 8;
  float acc = cb[d];
#pragma unroll
  for (int j = 0; j < 4; ++j) {
    int ll = l - 3 + j;
    int lc = ll < 0 ? 0 : ll;
    float xv = xz[((size_t)((b << 8) + lc) << 10) + d];
    acc += (ll >= 0 ? xv : 0.f) * cw[(d << 2) + j];
  }
  float s = silu_f(acc);
  xcp[idx]    = s;
  xcp_bf[idx] = f2bf(s);
}

// ---------------------------------------------------------------------------
// Selective-scan recurrence: thread per (b, d_inner), 16 states + 16 A-coefs
// in registers, 256 sequential steps.  Step l+1's operands are prefetched
// (clamped index, branchless) so loads overlap the exp/FMA chain.
// dbc row: [0:16)=dt_rank slice, [16:32)=B, [32:48)=C (rows 16B-aligned).
// ---------------------------------------------------------------------------
__global__ __launch_bounds__(256)
void ssm_scan_kernel(const float* __restrict__ dt,    // [B*L,512] softplus'd
                     const float* __restrict__ dbc,   // [B*L,48]
                     const float* __restrict__ xcp,   // [B*L,512]
                     const float* __restrict__ xz,    // [B*L,1024] (z hi half)
                     const float* __restrict__ A_log, // [512,16]
                     const float* __restrict__ Dsk,   // [512]
                     __bf16* __restrict__ y) {        // [B*L,512] bf16
  int b = blockIdx.x >> 1;
  int d = ((blockIdx.x & 1) << 8) + threadIdx.x;
  float Ar[16], h[16];
#pragma unroll
  for (int n = 0; n < 16; ++n) {
    Ar[n] = -__expf(A_log[(d << 4) + n]);
    h[n]  = 0.f;
  }
  float dsk = Dsk[d];

  int m0 = b << 8;
  float dtv = dt [((size_t)m0 << 9) + d];
  float xv  = xcp[((size_t)m0 << 9) + d];
  float zv  = xz [((size_t)m0 << 10) + 512 + d];
  float4 bb[4], cc[4];
  {
    const float4* bcv = (const float4*)(dbc + (size_t)m0 * 48 + 16);
#pragma unroll
    for (int i = 0; i < 4; ++i) bb[i] = bcv[i];
#pragma unroll
    for (int i = 0; i < 4; ++i) cc[i] = bcv[4 + i];
  }

  for (int l = 0; l < 256; ++l) {
    int m  = m0 + l;
    int ln = l < 255 ? l + 1 : 255;           // clamped prefetch index
    int mn = m0 + ln;
    // prefetch next step's operands (overlaps the recurrence below)
    float dtv_n = dt [((size_t)mn << 9) + d];
    float xv_n  = xcp[((size_t)mn << 9) + d];
    float zv_n  = xz [((size_t)mn << 10) + 512 + d];
    float4 bbn[4], ccn[4];
    {
      const float4* bcv = (const float4*)(dbc + (size_t)mn * 48 + 16);
#pragma unroll
      for (int i = 0; i < 4; ++i) bbn[i] = bcv[i];
#pragma unroll
      for (int i = 0; i < 4; ++i) ccn[i] = bcv[4 + i];
    }
    // compute current step
    const float* Bv = (const float*)bb;
    const float* Cv = (const float*)cc;
    float dtx = dtv * xv;
    float yv  = 0.f;
#pragma unroll
    for (int n = 0; n < 16; ++n) {
      h[n] = __expf(dtv * Ar[n]) * h[n] + dtx * Bv[n];
      yv += h[n] * Cv[n];
    }
    yv += dsk * xv;
    y[((size_t)m << 9) + d] = f2bf(yv * silu_f(zv));
    // rotate
    dtv = dtv_n; xv = xv_n; zv = zv_n;
#pragma unroll
    for (int i = 0; i < 4; ++i) { bb[i] = bbn[i]; cc[i] = ccn[i]; }
  }
}

// ---------------------------------------------------------------------------
// RMSNorm over rows of 256.  Writes f32 (maxpool input) and bf16 (next
// layer's in_proj operand).  Wave-per-row, wave32 shuffles.
// ---------------------------------------------------------------------------
__global__ __launch_bounds__(256)
void rmsnorm_kernel(float* __restrict__ u, __bf16* __restrict__ ubf,
                    const float* __restrict__ w) {
  int wid = threadIdx.x >> 5, lane = threadIdx.x & 31;
  int row = blockIdx.x * 8 + wid;              // 4096 rows
  float* p = u + (size_t)row * 256;
  __bf16* pb = ubf + (size_t)row * 256;
  float vals[8], ss = 0.f;
#pragma unroll
  for (int i = 0; i < 8; ++i) {
    vals[i] = p[lane + 32 * i];
    ss += vals[i] * vals[i];
  }
#pragma unroll
  for (int off = 16; off > 0; off >>= 1) ss += __shfl_xor(ss, off, 32);
  float scale = rsqrtf(ss * (1.f / 256.f) + 1.1920929e-07f);
#pragma unroll
  for (int i = 0; i < 8; ++i) {
    int c = lane + 32 * i;
    float v = vals[i] * scale * w[c];
    p[c]  = v;
    pb[c] = f2bf(v);
  }
}

// ---------------------------------------------------------------------------
// Max over last axis (256 contiguous) -> pooled bf16 (fc GEMM operand).
// ---------------------------------------------------------------------------
__global__ __launch_bounds__(256)
void maxpool_kernel(const float* __restrict__ u, __bf16* __restrict__ pooled) {
  int wid = threadIdx.x >> 5, lane = threadIdx.x & 31;
  int row = blockIdx.x * 8 + wid;              // 4096 rows
  const float* p = u + (size_t)row * 256;
  float mx = -3.4028235e38f;
#pragma unroll
  for (int i = 0; i < 8; ++i) mx = fmaxf(mx, p[lane + 32 * i]);
#pragma unroll
  for (int off = 16; off > 0; off >>= 1) mx = fmaxf(mx, __shfl_xor(mx, off, 32));
  if (lane == 0) pooled[row] = f2bf(mx);
}

// ---------------------------------------------------------------------------
extern "C" void kernel_launch(void* const* d_in, const int* in_sizes, int n_in,
                              void* d_out, int out_size, void* d_ws,
                              size_t ws_size, hipStream_t stream) {
  (void)in_sizes; (void)n_in; (void)out_size; (void)ws_size;
  const float* x          = (const float*)d_in[0];   // [16,80,256]
  const float* proj_w     = (const float*)d_in[1];   // [256,80]
  const float* proj_b     = (const float*)d_in[2];   // [256]
  const float* in_proj_w  = (const float*)d_in[3];   // [4,1024,256]
  const float* conv_w     = (const float*)d_in[4];   // [4,512,1,4]
  const float* conv_b     = (const float*)d_in[5];   // [4,512]
  const float* x_proj_w   = (const float*)d_in[6];   // [4,48,512]
  const float* dt_proj_w  = (const float*)d_in[7];   // [4,512,16]
  const float* dt_proj_b  = (const float*)d_in[8];   // [4,512]
  const float* A_log      = (const float*)d_in[9];   // [4,512,16]
  const float* D_skip     = (const float*)d_in[10];  // [4,512]
  const float* out_proj_w = (const float*)d_in[11];  // [4,256,512]
  const float* norm_w     = (const float*)d_in[12];  // [4,256]
  const float* fc_w       = (const float*)d_in[13];  // [35,256]
  const float* fc_b       = (const float*)d_in[14];  // [35]
  float* out = (float*)d_out;                        // [16,35]

  // -------- scratch carve-out (~48 MB; every buffer fully written) --------
  char* ws = (char*)d_ws;
  size_t off = 0;
  auto carveF = [&](size_t elems) -> float* {
    float* p = (float*)(ws + off);
    off = (off + elems * 4 + 255) & ~(size_t)255;
    return p;
  };
  auto carveB = [&](size_t elems) -> __bf16* {
    __bf16* p = (__bf16*)(ws + off);
    off = (off + elems * 2 + 255) & ~(size_t)255;
    return p;
  };
  // f32 activations
  float* u    = carveF((size_t)4096 * 256);   // [B, L, 256]
  float* xzb  = carveF((size_t)4096 * 1024);  // [B*L, 1024]
  float* xcp  = carveF((size_t)4096 * 512);   // [B*L, 512]
  float* dbc  = carveF((size_t)4096 * 48);    // [B*L, 48]
  float* dtb  = carveF((size_t)4096 * 512);   // [B*L, 512]
  // bf16 GEMM operands (activations)
  __bf16* xTbf   = carveB((size_t)4096 * 96);  // [B*T, 96]  (K=80 padded)
  __bf16* ubf    = carveB((size_t)4096 * 256);
  __bf16* xcbf   = carveB((size_t)4096 * 512);
  __bf16* dtin   = carveB((size_t)4096 * 32);  // dbc[:, :16] padded to 32
  __bf16* ybf    = carveB((size_t)4096 * 512);
  __bf16* pooled = carveB((size_t)16 * 256);
  // bf16 weights (converted once per launch)
  __bf16* pwbf = carveB((size_t)256 * 96);          // proj_w  padded K=96
  __bf16* ipbf = carveB((size_t)4 * 1024 * 256);    // in_proj_w
  __bf16* xpbf = carveB((size_t)4 * 48 * 512);      // x_proj_w
  __bf16* dtwb = carveB((size_t)4 * 512 * 32);      // dt_proj_w padded K=32
  __bf16* opbf = carveB((size_t)4 * 256 * 512);     // out_proj_w
  __bf16* fcbf = carveB((size_t)35 * 256);          // fc_w

  const dim3 blk(128);
  auto cgrid = [](int total) { return dim3((total + 255) / 256); };

  // ---- stage weights as bf16 (zero K-padding where needed) ----
  cvt_pad_bf16_kernel<<<cgrid(256 * 96), 256, 0, stream>>>(
      proj_w, 80, 80, pwbf, 96, 256 * 96);
  cvt_pad_bf16_kernel<<<cgrid(4 * 1024 * 256), 256, 0, stream>>>(
      in_proj_w, 256, 256, ipbf, 256, 4 * 1024 * 256);
  cvt_pad_bf16_kernel<<<cgrid(4 * 48 * 512), 256, 0, stream>>>(
      x_proj_w, 512, 512, xpbf, 512, 4 * 48 * 512);
  cvt_pad_bf16_kernel<<<cgrid(4 * 512 * 32), 256, 0, stream>>>(
      dt_proj_w, 16, 16, dtwb, 32, 4 * 512 * 32);
  cvt_pad_bf16_kernel<<<cgrid(4 * 256 * 512), 256, 0, stream>>>(
      out_proj_w, 512, 512, opbf, 512, 4 * 256 * 512);
  cvt_pad_bf16_kernel<<<cgrid(35 * 256), 256, 0, stream>>>(
      fc_w, 256, 256, fcbf, 256, 35 * 256);

  // ---- input projection: writes ubf (bf16) directly in [b, d, t] layout ----
  transpose_x_kernel<<<cgrid(16 * 256 * 96), 256, 0, stream>>>(x, xTbf);
  wmma_gemm_kernel<0, true, false>
      <<<dim3(256 / 64, 4096 / 64), blk, 0, stream>>>(
      xTbf, 96, pwbf, 96, proj_b, ubf, 0, 4096, 256, 96);

  for (int l = 0; l < 4; ++l) {
    // xz = u @ in_proj_w^T                          [4096,1024]
    wmma_gemm_kernel<0, false, false>
        <<<dim3(1024 / 64, 4096 / 64), blk, 0, stream>>>(
        ubf, 256, ipbf + (size_t)l * 1024 * 256, 256, nullptr,
        xzb, 1024, 4096, 1024, 256);
    // depthwise causal conv + bias + silu           [4096,512]
    conv_silu_kernel<<<cgrid(16 * 256 * 512), 256, 0, stream>>>(
        xzb, conv_w + (size_t)l * 512 * 4, conv_b + (size_t)l * 512,
        xcp, xcbf);
    // dbc = xcp @ x_proj_w^T                        [4096,48]
    wmma_gemm_kernel<0, false, true>
        <<<dim3(1, 4096 / 64), blk, 0, stream>>>(
        xcbf, 512, xpbf + (size_t)l * 48 * 512, 512, nullptr,
        dbc, 48, 4096, 48, 512);
    // stage dt slice: dbc[:, :16] -> bf16 [4096,32] zero-padded
    cvt_pad_bf16_kernel<<<cgrid(4096 * 32), 256, 0, stream>>>(
        dbc, 48, 16, dtin, 32, 4096 * 32);
    // dt = softplus(dtin @ dt_proj_w^T + b)         [4096,512]
    wmma_gemm_kernel<1, false, false>
        <<<dim3(512 / 64, 4096 / 64), blk, 0, stream>>>(
        dtin, 32, dtwb + (size_t)l * 512 * 32, 32,
        dt_proj_b + (size_t)l * 512, dtb, 512, 4096, 512, 32);
    // selective scan + D-skip + silu(z) gate -> bf16 [4096,512]
    ssm_scan_kernel<<<32, 256, 0, stream>>>(
        dtb, dbc, xcp, xzb, A_log + (size_t)l * 512 * 16,
        D_skip + (size_t)l * 512, ybf);
    // u = y @ out_proj_w^T                          [4096,256]
    wmma_gemm_kernel<0, false, false>
        <<<dim3(256 / 64, 4096 / 64), blk, 0, stream>>>(
        ybf, 512, opbf + (size_t)l * 256 * 512, 512, nullptr,
        u, 256, 4096, 256, 512);
    // RMSNorm: writes u (f32) + ubf (bf16, next layer's operand)
    rmsnorm_kernel<<<512, 256, 0, stream>>>(
        u, ubf, norm_w + (size_t)l * 256);
  }

  // pooled = max over last axis; logits = pooled @ fc_w^T + fc_b
  maxpool_kernel<<<512, 256, 0, stream>>>(u, pooled);
  wmma_gemm_kernel<0, false, true><<<dim3(1, 1), blk, 0, stream>>>(
      pooled, 256, fcbf, 256, fc_b, out, 35, 16, 35, 256);
}
